// PhimoeSparseMoeBlock_83897891160612
// MI455X (gfx1250) — matrix operand order repaired and checked by reading, not compiled, not run on hardware
//
#include <hip/hip_runtime.h>
#include <cstdint>
#include <cstddef>

#define H_DIM   4096
#define F_DIM   6400
#define NEXP    8
#define NTOK    512
#define MTILE   64                      // rows per (expert) segment tile
#define MAXROWS 1536                    // 1024 slots + per-expert pad to 64

typedef __attribute__((ext_vector_type(16))) __bf16 v16bf;
typedef __attribute__((ext_vector_type(8)))  __bf16 v8bf;
typedef __attribute__((ext_vector_type(8)))  float  v8f;

// ---------------- WMMA helpers (gfx1250 bf16 16x16x32, f32 accum) ----------

__device__ __forceinline__ v8f wmma_bf16(v16bf a, v16bf b, v8f c) {
  // (neg_a, A, neg_b, B, c_mod, C, reuse_a, reuse_b)
  return __builtin_amdgcn_wmma_f32_16x16x32_bf16(false, a, false, b, (short)0, c,
                                                 false, false);
}

// A-matrix 16x32 bf16 layout per lane: two runs of 8 consecutive K values:
//   K = base + 0..7   (vector elems 0..7)
//   K = base + 16..23 (vector elems 8..15)
// caller passes p = rowPtr + k0 + base, where base = (lane<16)?0:8.
__device__ __forceinline__ v16bf load_a16_from_f32(const float* __restrict__ p) {
  float4 f0 = *(const float4*)(p + 0);
  float4 f1 = *(const float4*)(p + 4);
  float4 f2 = *(const float4*)(p + 16);
  float4 f3 = *(const float4*)(p + 20);
  v16bf r;
  r[0]=(__bf16)f0.x;  r[1]=(__bf16)f0.y;  r[2]=(__bf16)f0.z;  r[3]=(__bf16)f0.w;
  r[4]=(__bf16)f1.x;  r[5]=(__bf16)f1.y;  r[6]=(__bf16)f1.z;  r[7]=(__bf16)f1.w;
  r[8]=(__bf16)f2.x;  r[9]=(__bf16)f2.y;  r[10]=(__bf16)f2.z; r[11]=(__bf16)f2.w;
  r[12]=(__bf16)f3.x; r[13]=(__bf16)f3.y; r[14]=(__bf16)f3.z; r[15]=(__bf16)f3.w;
  return r;
}

__device__ __forceinline__ v16bf load_a16_from_bf16(const __bf16* __restrict__ p) {
  v8bf lo = *(const v8bf*)(p);
  v8bf hi = *(const v8bf*)(p + 16);
  v16bf r;
#pragma unroll
  for (int j = 0; j < 8; ++j) { r[j] = lo[j]; r[8 + j] = hi[j]; }
  return r;
}

// B-matrix 32x16 bf16 layout per lane: K = base+0..7 and base+16..23,
// base = k0 + ((lane<16)?0:8); caller passes p = rowPtr + that offset.
__device__ __forceinline__ v16bf load_b16_from_f32(const float* __restrict__ p) {
  float4 f0 = *(const float4*)(p + 0);
  float4 f1 = *(const float4*)(p + 4);
  float4 f2 = *(const float4*)(p + 16);
  float4 f3 = *(const float4*)(p + 20);
  v16bf r;
  r[0]=(__bf16)f0.x;  r[1]=(__bf16)f0.y;  r[2]=(__bf16)f0.z;  r[3]=(__bf16)f0.w;
  r[4]=(__bf16)f1.x;  r[5]=(__bf16)f1.y;  r[6]=(__bf16)f1.z;  r[7]=(__bf16)f1.w;
  r[8]=(__bf16)f2.x;  r[9]=(__bf16)f2.y;  r[10]=(__bf16)f2.z; r[11]=(__bf16)f2.w;
  r[12]=(__bf16)f3.x; r[13]=(__bf16)f3.y; r[14]=(__bf16)f3.z; r[15]=(__bf16)f3.w;
  return r;
}

// ---------------- 1) Router: logits -> softmax -> top2 -> renormalize ------

__global__ void router_kernel(const float* __restrict__ x,
                              const float* __restrict__ gw,
                              int* __restrict__ topk_idx,
                              float* __restrict__ topk_w) {
  int t    = blockIdx.x;
  int wave = threadIdx.x >> 5;   // 8 waves = 8 experts
  int lane = threadIdx.x & 31;
  __shared__ float slog[NEXP];

  const float* xr = x + (size_t)t * H_DIM;
  const float* gr = gw + (size_t)wave * H_DIM;
  float partial = 0.f;
  for (int h = lane; h < H_DIM; h += 32) partial += xr[h] * gr[h];
#pragma unroll
  for (int off = 16; off; off >>= 1) partial += __shfl_down(partial, off, 32);
  if (lane == 0) slog[wave] = partial;
  __syncthreads();

  if (threadIdx.x == 0) {
    float mx = slog[0];
#pragma unroll
    for (int e = 1; e < NEXP; ++e) mx = fmaxf(mx, slog[e]);
    float p[NEXP]; float sum = 0.f;
#pragma unroll
    for (int e = 0; e < NEXP; ++e) { p[e] = __expf(slog[e] - mx); sum += p[e]; }
#pragma unroll
    for (int e = 0; e < NEXP; ++e) p[e] /= sum;
    int i0 = 0;
#pragma unroll
    for (int e = 1; e < NEXP; ++e) if (p[e] > p[i0]) i0 = e;
    int i1 = (i0 == 0) ? 1 : 0;
#pragma unroll
    for (int e = 0; e < NEXP; ++e) if (e != i0 && p[e] > p[i1]) i1 = e;
    float s = p[i0] + p[i1];
    topk_idx[t * 2 + 0] = i0;        topk_idx[t * 2 + 1] = i1;
    topk_w[t * 2 + 0]   = p[i0] / s; topk_w[t * 2 + 1]   = p[i1] / s;
  }
}

// ---------------- 2) Deterministic expert grouping (pad to MTILE rows) -----

__global__ void build_perm_kernel(const int* __restrict__ topk_idx,
                                  int* __restrict__ perm,
                                  int* __restrict__ slot_of,
                                  int* __restrict__ seg) {
  // seg[e] = row base, seg[NEXP+e] = #MTILE-row tiles
  int base = 0;
  for (int e = 0; e < NEXP; ++e) {
    seg[e] = base;
    int cnt = 0;
    for (int t = 0; t < NTOK; ++t)
      for (int k = 0; k < 2; ++k)
        if (topk_idx[t * 2 + k] == e) {
          perm[base + cnt] = t;
          slot_of[t * 2 + k] = base + cnt;
          cnt++;
        }
    int padded = (cnt + (MTILE - 1)) & ~(MTILE - 1);
    for (int i = cnt; i < padded; ++i) perm[base + i] = 0;  // safe dummy token
    seg[NEXP + e] = padded / MTILE;
    base += padded;
  }
}

// ---------------- 3) GEMM1: act = silu(Xg @ w1^T) * (Xg @ w3^T) ------------
// Block tile: 64 rows x 128 F-cols; 8 waves; each wave M=64 (4 A tiles),
// N=16, K over H. B tiles (w1,w3) amortized over 4 M tiles -> 8 WMMA/step.

__global__ void gemm1_kernel(const float* __restrict__ x,
                             const float* __restrict__ w1,
                             const float* __restrict__ w3,
                             const int* __restrict__ perm,
                             const int* __restrict__ seg,
                             __bf16* __restrict__ act) {
  int e = blockIdx.z;
  if ((int)blockIdx.y >= seg[NEXP + e]) return;
  int rowBase = seg[e] + (int)blockIdx.y * MTILE;
  int wave = threadIdx.x >> 5, lane = threadIdx.x & 31;
  int fBase = (int)blockIdx.x * 128 + wave * 16;
  int mn = lane & 15;                       // lane's M row (A) and N col (B)

  const float* xr[4];
#pragma unroll
  for (int m = 0; m < 4; ++m) {
    int tok = perm[rowBase + m * 16 + mn];
    xr[m] = x + (size_t)tok * H_DIM;
  }
  const float* w1r = w1 + ((size_t)e * F_DIM + fBase + mn) * (size_t)H_DIM;
  const float* w3r = w3 + ((size_t)e * F_DIM + fBase + mn) * (size_t)H_DIM;
  int aoff = (lane < 16) ? 0 : 8;
  int boff = (lane < 16) ? 0 : 8;

  v8f acc1[4] = {{}, {}, {}, {}};
  v8f acc3[4] = {{}, {}, {}, {}};
  for (int k0 = 0; k0 < H_DIM; k0 += 32) {
    v16bf b1 = load_b16_from_f32(w1r + k0 + boff);
    v16bf b3 = load_b16_from_f32(w3r + k0 + boff);
#pragma unroll
    for (int m = 0; m < 4; ++m) {
      v16bf a = load_a16_from_f32(xr[m] + k0 + aoff);
      acc1[m] = wmma_bf16(a, b1, acc1[m]);
      acc3[m] = wmma_bf16(a, b3, acc3[m]);
    }
  }

  int mb = (lane < 16) ? 0 : 8;
#pragma unroll
  for (int m = 0; m < 4; ++m) {
#pragma unroll
    for (int r = 0; r < 8; ++r) {
      int row = rowBase + m * 16 + mb + r;
      float g = acc1[m][r], u = acc3[m][r];
      float sig = __fdividef(1.f, 1.f + __expf(-g));   // rcp-based sigmoid
      act[(size_t)row * F_DIM + fBase + mn] = (__bf16)(g * sig * u);
    }
  }
}

// ---------------- 4) GEMM2: g2 = act @ w2^T (unscaled, per row-slot) -------

__global__ void gemm2_kernel(const __bf16* __restrict__ act,
                             const float* __restrict__ w2,
                             const int* __restrict__ seg,
                             float* __restrict__ g2) {
  int e = blockIdx.z;
  if ((int)blockIdx.y >= seg[NEXP + e]) return;
  int rowBase = seg[e] + (int)blockIdx.y * MTILE;
  int wave = threadIdx.x >> 5, lane = threadIdx.x & 31;
  int hBase = (int)blockIdx.x * 128 + wave * 16;
  int mn = lane & 15;

  const __bf16* ar[4];
#pragma unroll
  for (int m = 0; m < 4; ++m)
    ar[m] = act + (size_t)(rowBase + m * 16 + mn) * F_DIM;
  const float* w2r = w2 + ((size_t)e * H_DIM + hBase + mn) * (size_t)F_DIM;
  int aoff = (lane < 16) ? 0 : 8;
  int boff = (lane < 16) ? 0 : 8;

  v8f acc[4] = {{}, {}, {}, {}};
  for (int k0 = 0; k0 < F_DIM; k0 += 32) {
    v16bf b = load_b16_from_f32(w2r + k0 + boff);
#pragma unroll
    for (int m = 0; m < 4; ++m) {
      v16bf a = load_a16_from_bf16(ar[m] + k0 + aoff);
      acc[m] = wmma_bf16(a, b, acc[m]);
    }
  }

  int mb = (lane < 16) ? 0 : 8;
#pragma unroll
  for (int m = 0; m < 4; ++m) {
#pragma unroll
    for (int r = 0; r < 8; ++r) {
      int row = rowBase + m * 16 + mb + r;
      g2[(size_t)row * H_DIM + hBase + mn] = acc[m][r];
    }
  }
}

// ---------------- 5) Combine: out[t] = w0*g2[slot0] + w1*g2[slot1] ---------

__global__ void combine_kernel(const float* __restrict__ g2,
                               const int* __restrict__ slot_of,
                               const float* __restrict__ topk_w,
                               float* __restrict__ out) {
  int t = blockIdx.x;
  int s0 = slot_of[t * 2 + 0], s1 = slot_of[t * 2 + 1];
  float a = topk_w[t * 2 + 0], b = topk_w[t * 2 + 1];
  const float* r0 = g2 + (size_t)s0 * H_DIM;
  const float* r1 = g2 + (size_t)s1 * H_DIM;
  float* o = out + (size_t)t * H_DIM;
  for (int h = threadIdx.x; h < H_DIM; h += blockDim.x)
    o[h] = a * r0[h] + b * r1[h];
}

// ---------------- host launcher --------------------------------------------

extern "C" void kernel_launch(void* const* d_in, const int* in_sizes, int n_in,
                              void* d_out, int out_size, void* d_ws, size_t ws_size,
                              hipStream_t stream) {
  (void)in_sizes; (void)n_in; (void)out_size; (void)ws_size;
  const float* x  = (const float*)d_in[0];
  const float* gw = (const float*)d_in[1];
  const float* w1 = (const float*)d_in[2];
  const float* w2 = (const float*)d_in[3];   // dict order: x, gate_w, w1, w2, w3
  const float* w3 = (const float*)d_in[4];
  float* out = (float*)d_out;

  char*  ws  = (char*)d_ws;
  size_t off = 0;
  auto take = [&](size_t bytes) -> char* {
    char* p = ws + off;
    off = (off + bytes + 255) & ~(size_t)255;
    return p;
  };
  int*    topk_idx = (int*)  take((size_t)NTOK * 2 * sizeof(int));
  float*  topk_w   = (float*)take((size_t)NTOK * 2 * sizeof(float));
  int*    perm     = (int*)  take((size_t)MAXROWS * sizeof(int));
  int*    slot_of  = (int*)  take((size_t)NTOK * 2 * sizeof(int));
  int*    seg      = (int*)  take((size_t)2 * NEXP * sizeof(int));
  __bf16* act      = (__bf16*)take((size_t)MAXROWS * F_DIM * sizeof(uint16_t));
  float*  g2       = (float*) take((size_t)MAXROWS * H_DIM * sizeof(float));

  router_kernel<<<NTOK, 256, 0, stream>>>(x, gw, topk_idx, topk_w);
  build_perm_kernel<<<1, 1, 0, stream>>>(topk_idx, perm, slot_of, seg);
  gemm1_kernel<<<dim3(F_DIM / 128, NTOK / MTILE, NEXP), 256, 0, stream>>>(
      x, w1, w3, perm, seg, act);
  gemm2_kernel<<<dim3(H_DIM / 128, NTOK / MTILE, NEXP), 256, 0, stream>>>(
      act, w2, seg, g2);
  combine_kernel<<<NTOK, 256, 0, stream>>>(g2, slot_of, topk_w, out);
}
